// GCN_84499186582104
// MI455X (gfx1250) — matrix operand order repaired and checked by reading, not compiled, not run on hardware
//
#include <hip/hip_runtime.h>

typedef __attribute__((ext_vector_type(16))) _Float16 v16h;
typedef __attribute__((ext_vector_type(8)))  float    v8f;

// bias/small-weight LDS offsets (floats)
#define OBB 0      // bb   8
#define OB1 8      // b1   16
#define OB2 24     // b2   8
#define OBG 32     // bg   8
#define OWF 40     // wf   2x8
#define OBF 56     // bf   2

// per-wave buffer layout (floats); tile pitch 20 dwords = 80B (16B-aligned rows)
#define PITCH 20
#define OAW 0      // 128 : adj pair
#define OEW 128    // 128 : edge_attr pair
#define OTL 256    // 320 : 16 x PITCH f32 tile
#define WBUF 576   // 2304B per wave (16B multiple)

// A-fragment (16x32 f16) from 16xPITCH LDS tile; K in [0,16) from tile, K>=kmax -> 0.
// ISA layout: lane L holds row M=L%16; lanes<16: K={0..7,16..23}, lanes>=16: K={8..15,24..31}.
// Our tiles never have K>=16 data, so elements 8..15 are always zero; the lane's 8 live
// K-slots are the 8 contiguous floats at row*PITCH + 8*hi  ->  two ds_load_b128.
__device__ __forceinline__ v16h a_frag_tile(const float* t, int lane, int kmax) {
  const int M = lane & 15, hi = lane >> 4;
  const float4* r = (const float4*)(t + M * PITCH + (hi ? 8 : 0));
  const float4 q0 = r[0], q1 = r[1];
  const float q[8] = {q0.x, q0.y, q0.z, q0.w, q1.x, q1.y, q1.z, q1.w};
  v16h a;
#pragma unroll
  for (int e = 0; e < 8; ++e) {
    const int K = e + (hi ? 8 : 0);
    a[e]     = (K < kmax) ? (_Float16)q[e] : (_Float16)0.0f;
    a[e + 8] = (_Float16)0.0f;               // K = 16..31 always zero here
  }
  return a;
}

// Convert previous D (16x16, lane=col N, half=row-group) into B (32x16, rows 16..31 = 0).
// B layout: lane holds col N=L%16; lanes<16: K={0..7,16..23}, lanes>=16: K={8..15,24..31}.
__device__ __forceinline__ v16h b_frag_from_d(v8f d, int lane) {
  const int hi = lane >> 4;
  float o[8];
#pragma unroll
  for (int i = 0; i < 8; ++i) o[i] = __shfl_xor(d[i], 16, 32);
  v16h b;
#pragma unroll
  for (int e = 0; e < 16; ++e) {
    const float v = (e < 8) ? d[e] : o[e - 8];
    b[e] = (e < 8) ? (_Float16)v : (_Float16)0.0f;
  }
  return b;
}

__global__ __launch_bounds__(256) void gcn_fused_wmma(
    const float* __restrict__ x,  const float* __restrict__ adj,
    const float* __restrict__ ea, const float* __restrict__ wb,
    const float* __restrict__ bb, const float* __restrict__ w1,
    const float* __restrict__ b1, const float* __restrict__ w2,
    const float* __restrict__ b2, const float* __restrict__ wg,
    const float* __restrict__ bg, const float* __restrict__ wf,
    const float* __restrict__ bf, float* __restrict__ out, int npairs)
{
  __shared__ v16h  wfr[4][32];                 // pre-packed f16 weight B-fragments
  __shared__ float bias[64];                   // bb,b1,b2,bg,wf,bf
  __shared__ __align__(16) float wbuf[8][WBUF];

  const int tid  = threadIdx.x;
  const int lane = tid & 31;
  const int wid  = tid >> 5;
  int p = blockIdx.x * 8 + wid;
  if (p >= npairs) p = npairs - 1;             // uniform barriers; duplicate work is benign

  // ---- build the 4 weight B-fragments ONCE per block (threads 0..127, one slot each) ----
  if (tid < 128) {
    const int f = tid >> 5, l = tid & 31;
    const int N = l & 15, h2 = l >> 4;
    v16h v;
#pragma unroll
    for (int e = 0; e < 16; ++e) {
      const int K = (e < 8) ? (e + 8 * h2) : (16 + 8 * h2 + (e - 8));
      float s;
      if      (f == 0) s = (N < 8)           ? wb[N * 32 + K] : 0.f;  // B[k,n]=wb[n,k]
      else if (f == 1) s =                     w1[K * 16 + N];        // B[k,n]=w1[k,n]
      else if (f == 2) s = (K < 16 && N < 8) ? w2[K * 8 + N] : 0.f;   // B[k,n]=w2[k,n]
      else             s = (K < 8  && N < 8) ? wg[N * 8 + K] : 0.f;   // B[k,n]=wg[n,k]
      v[e] = (_Float16)s;
    }
    wfr[f][l] = v;
  }
  if (tid < 8)       bias[OBB + tid]      = bb[tid];
  else if (tid < 24) bias[OB1 + tid - 8]  = b1[tid - 8];
  else if (tid < 32) bias[OB2 + tid - 24] = b2[tid - 24];
  else if (tid < 40) bias[OBG + tid - 32] = bg[tid - 32];
  else if (tid < 56) bias[OWF + tid - 40] = wf[tid - 40];
  else if (tid < 58) bias[OBF + tid - 56] = bf[tid - 56];

  float* aw = &wbuf[wid][OAW];
  float* ew = &wbuf[wid][OEW];
  float* tl = &wbuf[wid][OTL];

  // ---- stage adj / edge_attr pair (coalesced; consumed transposed later) ----
  const float* ag = adj + (size_t)p * 128;
  const float* eg = ea  + (size_t)p * 128;
#pragma unroll
  for (int i = 0; i < 4; ++i) { aw[lane + i*32] = ag[lane + i*32]; ew[lane + i*32] = eg[lane + i*32]; }

  // ---- feat A-fragment straight from global (lane owns one stacked row) ----
  const int n16 = lane & 15, hi = lane >> 4;
  const float* xrow = x + ((size_t)p * 16 + n16) * 33;   // row M = lane%16 of the pair
  const float tyM = xrow[0];                              // node type of this row
  v16h aF;
  {
    const int kb0 = hi ? 8 : 0, kb1 = hi ? 24 : 16;
#pragma unroll
    for (int e = 0; e < 8; ++e)  aF[e]     = (_Float16)xrow[1 + kb0 + e];
#pragma unroll
    for (int e = 0; e < 8; ++e)  aF[8 + e] = (_Float16)xrow[1 + kb1 + e];
  }
  float tyR[8];                                           // types of this lane's D-frag rows
#pragma unroll
  for (int i = 0; i < 8; ++i) tyR[i] = __shfl(tyM, 8 * hi + i, 32);

  __syncthreads();

  const v8f cz = {};
  const v16h bWB = wfr[0][lane];
  const v16h bW1 = wfr[1][lane];

  // ---- 1) Fk = feat·wbT + bb ;  P = feat·w1   (K=32, one WMMA each) ----
  v8f dfk = __builtin_amdgcn_wmma_f32_16x16x32_f16(false, aF, false, bWB, (short)0, cz, false, false);
  v8f dp  = __builtin_amdgcn_wmma_f32_16x16x32_f16(false, aF, false, bW1, (short)0, cz, false, false);
  const float bbn = bias[OBB + (n16 & 7)];
#pragma unroll
  for (int i = 0; i < 8; ++i) dfk[i] += bbn;

  // stash Fk (stacked rows=(b,v), cols=o) into tile: immediate offsets i*PITCH
#pragma unroll
  for (int i = 0; i < 8; ++i) tl[(8*hi + i)*PITCH + n16] = dfk[i];
  __syncthreads();

  // ---- 2) M2 = Fk·Fk per batch, column softmax, W = adj*(S+ea)  (VALU, in-lane) ----
  const int bl = hi, j = n16 & 7;              // this lane owns column j of batch bl
  const float* blk = tl + 8 * bl * PITCH;      // this batch's 8x8 Fk block (rows 16B-aligned)
  float rj[8];                                 // Fk[b,k,j] = block row j (contiguous)
  {
    const float4* rp = (const float4*)(blk + j * PITCH);
    const float4 r0 = rp[0], r1 = rp[1];
    rj[0]=r0.x; rj[1]=r0.y; rj[2]=r0.z; rj[3]=r0.w; rj[4]=r1.x; rj[5]=r1.y; rj[6]=r1.z; rj[7]=r1.w;
  }
  float m2[8] = {0.f,0.f,0.f,0.f,0.f,0.f,0.f,0.f};
#pragma unroll
  for (int k = 0; k < 8; ++k) {                // block row k = Fk[b,i,k] over i (ds_load_b128 x2)
    const float4* rp = (const float4*)(blk + k * PITCH);
    const float4 r0 = rp[0], r1 = rp[1];
    const float rw[8] = {r0.x, r0.y, r0.z, r0.w, r1.x, r1.y, r1.z, r1.w};
#pragma unroll
    for (int i = 0; i < 8; ++i) m2[i] += rw[i] * rj[k];
  }
  float mx = m2[0];
#pragma unroll
  for (int i = 1; i < 8; ++i) mx = fmaxf(mx, m2[i]);
  float se = 0.f;
#pragma unroll
  for (int i = 0; i < 8; ++i) { m2[i] = __expf(m2[i] - mx); se += m2[i]; }
  const float rse = 1.0f / se;
  float wcol[8];
#pragma unroll
  for (int i = 0; i < 8; ++i)
    wcol[i] = aw[bl*64 + i*8 + j] * (m2[i]*rse + ew[bl*64 + i*8 + j]);
  __syncthreads();

  // block-diagonal W into tile (zero first)
#pragma unroll
  for (int i = 0; i < 8; ++i) tl[(8*hi + i)*PITCH + n16] = 0.f;
  __syncthreads();
  if (n16 < 8) {
#pragma unroll
    for (int i = 0; i < 8; ++i) tl[(8*bl + i)*PITCH + (8*bl + j)] = wcol[i];
  }
  __syncthreads();

  // ---- 3) H1 = leaky(Wblk · P + b1) ----
  const v16h aW = a_frag_tile(tl, lane, 16);   // W operand kept in registers for reuse
  v8f dh1 = __builtin_amdgcn_wmma_f32_16x16x32_f16(false, aW, false, b_frag_from_d(dp, lane), (short)0, cz, false, false);
  const float b1n = bias[OB1 + n16];
#pragma unroll
  for (int i = 0; i < 8; ++i) { float v = dh1[i] + b1n; dh1[i] = v > 0.f ? v : 0.01f * v; }

  // transpose H1 via tile -> A-fragment
  __syncthreads();
#pragma unroll
  for (int i = 0; i < 8; ++i) tl[(8*hi + i)*PITCH + n16] = dh1[i];
  __syncthreads();
  const v16h aH1 = a_frag_tile(tl, lane, 16);

  // ---- 4) Q = H1·w2 ; H2 = leaky(Wblk·Q + b2) ----
  v8f dq  = __builtin_amdgcn_wmma_f32_16x16x32_f16(false, aH1, false, wfr[2][lane], (short)0, cz, false, false);
  v8f dh2 = __builtin_amdgcn_wmma_f32_16x16x32_f16(false, aW,  false, b_frag_from_d(dq, lane), (short)0, cz, false, false);
  const float b2n = bias[OB2 + (n16 & 7)];
#pragma unroll
  for (int i = 0; i < 8; ++i) { float v = dh2[i] + b2n; dh2[i] = v > 0.f ? v : 0.01f * v; }

  // transpose H2 via tile -> A-fragment (only K<8 meaningful)
  __syncthreads();
#pragma unroll
  for (int i = 0; i < 8; ++i) tl[(8*hi + i)*PITCH + n16] = dh2[i];
  __syncthreads();
  const v16h aH2 = a_frag_tile(tl, lane, 8);

  // ---- 5) two masked attention pools: gate = softmax_v(hm·wgT + bg); pool = sum_v gate*hm ----
  const float bgn = bias[OBG + (n16 & 7)];
  const v16h bWG = wfr[3][lane];
  const v16h zh = {};
  float xpool = 0.f;
#pragma unroll
  for (int t = 0; t < 2; ++t) {
    const float tf = (float)t;
    const v16h am = (tyM == tf) ? aH2 : zh;    // row mask folded into A operand
    v8f dg = __builtin_amdgcn_wmma_f32_16x16x32_f16(false, am, false, bWG, (short)0, cz, false, false);
    float gl[8];
    float gmx = dg[0] + bgn;
#pragma unroll
    for (int i = 0; i < 8; ++i) { gl[i] = dg[i] + bgn; gmx = fmaxf(gmx, gl[i]); }
    float gs = 0.f;
#pragma unroll
    for (int i = 0; i < 8; ++i) { gl[i] = __expf(gl[i] - gmx); gs += gl[i]; }
    const float rg = 1.0f / gs;
    float pool = 0.f;
#pragma unroll
    for (int i = 0; i < 8; ++i)
      pool += gl[i] * rg * dh2[i] * ((tyR[i] == tf) ? 1.f : 0.f);  // gate * (h*mask)
    xpool += 0.5f * pool;                      // mean over the two pools
  }

  // ---- 6) final linear out[2p+b, c] = xpool·wf[c,:] + bf[c] ----
  __syncthreads();
  if (n16 < 8) tl[hi * 8 + n16] = xpool;       // tile reused as [2][8] scratch
  __syncthreads();
  if (lane < 4) {
    const int b = lane >> 1, c = lane & 1;
    float acc = bias[OBF + c];
#pragma unroll
    for (int o = 0; o < 8; ++o) acc += tl[b * 8 + o] * bias[OWF + c * 8 + o];
    out[(size_t)p * 4 + b * 2 + c] = acc;
  }
}

extern "C" void kernel_launch(void* const* d_in, const int* in_sizes, int n_in,
                              void* d_out, int out_size, void* d_ws, size_t ws_size,
                              hipStream_t stream) {
  (void)n_in; (void)d_ws; (void)ws_size; (void)out_size;
  const float* x   = (const float*)d_in[0];
  const float* adj = (const float*)d_in[1];
  const float* ea  = (const float*)d_in[2];
  // d_in[3]=wa, d_in[4]=ba are unused by the reference
  const float* wb  = (const float*)d_in[5];
  const float* bb  = (const float*)d_in[6];
  const float* w1  = (const float*)d_in[7];
  const float* b1  = (const float*)d_in[8];
  const float* w2  = (const float*)d_in[9];
  const float* b2  = (const float*)d_in[10];
  const float* wg  = (const float*)d_in[11];
  const float* bg  = (const float*)d_in[12];
  const float* wf  = (const float*)d_in[13];
  const float* bf  = (const float*)d_in[14];
  float* out = (float*)d_out;

  const int B = in_sizes[0] / (8 * 33);   // x is (B, 8, 33)
  const int npairs = B / 2;               // one wave per 2 batches
  const int blocks = (npairs + 7) / 8;    // 8 waves (256 threads) per block

  gcn_fused_wmma<<<blocks, 256, 0, stream>>>(x, adj, ea, wb, bb, w1, b1,
                                             w2, b2, wg, bg, wf, bf, out, npairs);
}
